// NLSPN_71605694759613
// MI455X (gfx1250) — matrix operand order repaired and checked by reading, not compiled, not run on hardware
//
#include <hip/hip_runtime.h>
#include <math.h>

// Problem constants (from reference)
#define BB   2
#define CG   64
#define HH   240
#define WW   1216
#define NUMD 8
#define PROP 6

typedef __attribute__((ext_vector_type(16))) _Float16 v16h;
typedef __attribute__((ext_vector_type(8)))  _Float16 v8h;
typedef __attribute__((ext_vector_type(8)))  float    v8f;

union V16 { v16h v; v8h h[2]; };

// ---------------------------------------------------------------------------
// Kernel 0: pad/convert conv weights (24,64,3,3) f32 -> Wp[32][64][16] f16
// (M padded 24->32, K-per-channel padded 9->16 so one WMMA k-step = 2 channels)
// ---------------------------------------------------------------------------
__global__ void prep_weights(const float* __restrict__ w, _Float16* __restrict__ wp) {
    int i = blockIdx.x * 256 + threadIdx.x;            // 32*64*16 = 32768
    if (i >= 32 * 64 * 16) return;
    int r = i & 15;
    int c = (i >> 4) & 63;
    int m = i >> 10;
    float val = (m < 24 && r < 9) ? w[(m * 64 + c) * 9 + r] : 0.0f;
    wp[i] = (_Float16)val;
}

// ---------------------------------------------------------------------------
// Kernel 1: conv3x3 as implicit GEMM with V_WMMA_F32_16X16X32_F16.
// Block = 64 threads = 2 waves; each wave computes TWO 16-pixel N-tiles
// (32 px) x 24 channels, sharing the A (weight) fragments between tiles:
// per k-step = 4 b128 weight loads + 18 tap loads + 4 WMMA for 32 pixels.
// ---------------------------------------------------------------------------
__global__ __launch_bounds__(64) void conv_wmma(
    const float* __restrict__ g,        // guidance (B,64,H,W) f32
    const _Float16* __restrict__ wp,    // padded weights [32][64][16]
    const float* __restrict__ bias,     // conv_b (24)
    const float* __restrict__ scale,    // aff_scale_const (1)
    float* __restrict__ out_off,        // (B,18,H,W) region of d_out
    float* __restrict__ aff_pre)        // (B,8,H,W) workspace
{
    const int lane = threadIdx.x & 31;
    const int wave = threadIdx.x >> 5;
    const int n    = lane & 15;     // pixel column within tile / A-row
    const int half = lane >> 4;     // K-half selector
    const int w0   = (blockIdx.x * 2 + wave) * 32;
    const int wA   = w0 + n;        // N-tile A pixel
    const int wB   = w0 + 16 + n;   // N-tile B pixel
    const int h    = blockIdx.y;
    const int b    = blockIdx.z;
    const size_t P1 = (size_t)HH * WW;

    v8f acc0A = {};   // out channels 0..15 (offsets), pixels wA
    v8f acc1A = {};   // out channels 16..23 (affinity), pixels wA
    v8f acc0B = {};   // pixels wB
    v8f acc1B = {};

    for (int kb = 0; kb < 32; ++kb) {
        const int c0  = kb * 2;
        const int cme = c0 + half;
        const float* gc = g + (size_t)(b * CG + cme) * P1;

        // Prefetch the tap row two k-steps ahead (channel cme+4 < 64 when kb<30).
        if (kb < 30)
            __builtin_prefetch(gc + 4 * P1 + (size_t)h * WW + wA, 0, 1);

        // ---- B fragments: 9 conv taps of channel cme at each tile's pixel ----
        V16 bfA, bfB;
        #pragma unroll
        for (int i = 0; i < 16; ++i) { bfA.v[i] = (_Float16)0.0f; bfB.v[i] = (_Float16)0.0f; }
        #pragma unroll
        for (int r = 0; r < 9; ++r) {
            const int dy = r / 3 - 1, dx = r % 3 - 1;
            const int yy = h + dy;
            const bool yok = (yy >= 0 && yy < HH);
            const int xA = wA + dx, xB = wB + dx;
            float vA = 0.0f, vB = 0.0f;
            if (yok && xA >= 0 && xA < WW) vA = gc[(size_t)yy * WW + xA];
            if (yok && xB >= 0 && xB < WW) vB = gc[(size_t)yy * WW + xB];
            bfA.v[r] = (_Float16)vA;
            bfB.v[r] = (_Float16)vB;
        }

        // ---- A fragments: contiguous 8-half slices of padded weights ----
        const _Float16* p0 = wp + (((size_t)n        * 64 + cme) * 16);
        const _Float16* p1 = wp + (((size_t)(n + 16) * 64 + cme) * 16);
        V16 a0, a1;
        a0.h[0] = *(const v8h*)(p0);        // channel cme, r 0..7 (M rows 0..15)
        a0.h[1] = *(const v8h*)(p0 + 8);    // channel cme, r 8..15
        a1.h[0] = *(const v8h*)(p1);        // M rows 16..31
        a1.h[1] = *(const v8h*)(p1 + 8);

        acc0A = __builtin_amdgcn_wmma_f32_16x16x32_f16(
                    false, a0.v, false, bfA.v, (short)0, acc0A, false, false);
        acc1A = __builtin_amdgcn_wmma_f32_16x16x32_f16(
                    false, a1.v, false, bfA.v, (short)0, acc1A, false, false);
        acc0B = __builtin_amdgcn_wmma_f32_16x16x32_f16(
                    false, a0.v, false, bfB.v, (short)0, acc0B, false, false);
        acc1B = __builtin_amdgcn_wmma_f32_16x16x32_f16(
                    false, a1.v, false, bfB.v, (short)0, acc1B, false, false);
    }

    // ---- Epilogue (both pixel tiles) ----
    const float sc = 1.0f / (scale[0] + 1e-8f);
    const size_t pixA = (size_t)h * WW + wA;
    const size_t pixB = (size_t)h * WW + wB;
    #pragma unroll
    for (int v = 0; v < 8; ++v) {
        const int m = v + half * 8;                 // oa channel of acc0*[v]
        const int ch = (m < 8) ? m : m + 2;         // skip inserted ref channels 8,9
        out_off[(size_t)(b * 18 + ch) * P1 + pixA] = acc0A[v] + bias[m];
        out_off[(size_t)(b * 18 + ch) * P1 + pixB] = acc0B[v] + bias[m];
        if (half == 0) {
            const float bb = bias[16 + v];
            aff_pre[(size_t)(b * 8 + v) * P1 + pixA] = tanhf(acc1A[v] + bb) * sc;
            aff_pre[(size_t)(b * 8 + v) * P1 + pixB] = tanhf(acc1B[v] + bb) * sc;
            if (v < 2) {  // zero reference-tap offset channels 8 (y) and 9 (x)
                out_off[(size_t)(b * 18 + 8 + v) * P1 + pixA] = 0.0f;
                out_off[(size_t)(b * 18 + 8 + v) * P1 + pixB] = 0.0f;
            }
        }
    }
}

// ---------------------------------------------------------------------------
// Bilinear sample with zero padding outside (matches reference semantics)
// ---------------------------------------------------------------------------
__device__ __forceinline__ float bilin(const float* __restrict__ img, float y, float x) {
    const float y0 = floorf(y), x0 = floorf(x);
    const float wy1 = y - y0, wx1 = x - x0;
    float out = 0.0f;
    #pragma unroll
    for (int i = 0; i < 2; ++i) {
        #pragma unroll
        for (int j = 0; j < 2; ++j) {
            const float yy = y0 + (float)i;
            const float xx = x0 + (float)j;
            const float wy = i ? wy1 : 1.0f - wy1;
            const float wx = j ? wx1 : 1.0f - wx1;
            const bool valid = (yy >= 0.0f) && (yy <= (float)(HH - 1)) &&
                               (xx >= 0.0f) && (xx <= (float)(WW - 1));
            const int yi = (int)fminf(fmaxf(yy, 0.0f), (float)(HH - 1));
            const int xi = (int)fminf(fmaxf(xx, 0.0f), (float)(WW - 1));
            const float v = valid ? img[(size_t)yi * WW + xi] : 0.0f;
            out += wy * wx * v;
        }
    }
    return out;
}

// ---------------------------------------------------------------------------
// Kernel 2: affinity finalize (confidence sampling + L1 normalization)
// ---------------------------------------------------------------------------
__global__ void aff_finalize(
    const float* __restrict__ conf,     // (B,1,H,W)
    const float* __restrict__ off,      // (B,18,H,W) in d_out
    const float* __restrict__ aff_pre,  // (B,8,H,W) ws
    const float* __restrict__ scale,
    float* __restrict__ aff_full,       // (B,9,H,W) region of d_out
    float* __restrict__ out_scale)      // scalar passthrough
{
    const size_t P1 = (size_t)HH * WW;
    const size_t idx = (size_t)blockIdx.x * 256 + threadIdx.x;
    if (idx >= (size_t)BB * P1) return;
    const int b = (int)(idx / P1);
    const size_t pix = idx % P1;
    const int h = (int)(pix / WW);
    const int w = (int)(pix % WW);
    const float* cimg = conf + (size_t)b * P1;

    const int nb[8] = {0, 1, 2, 3, 5, 6, 7, 8};
    float a[8];
    float s = 0.0f;
    #pragma unroll
    for (int t = 0; t < 8; ++t) {
        const int k = nb[t];
        const float oy = off[(size_t)(b * 18 + 2 * k)     * P1 + pix];
        const float ox = off[(size_t)(b * 18 + 2 * k + 1) * P1 + pix];
        const float cs = bilin(cimg, (float)h + oy, (float)w + ox);
        const float av = aff_pre[(size_t)(b * 8 + t) * P1 + pix] * cs;
        a[t] = av;
        s += fabsf(av);
    }
    s += 1e-5f;
    if (s < 1.0f) s = 1.0f;
    float sum = 0.0f;
    #pragma unroll
    for (int t = 0; t < 8; ++t) { a[t] /= s; sum += a[t]; }
    const float aref = 1.0f - sum;
    #pragma unroll
    for (int k = 0; k < 9; ++k) {
        const float v = (k == 4) ? aref : a[k - (k > 4 ? 1 : 0)];
        aff_full[(size_t)(b * 9 + k) * P1 + pix] = v;
    }
    if (idx == 0) out_scale[0] = scale[0];
}

// ---------------------------------------------------------------------------
// Kernel 3: one propagation step (9-tap deformable bilinear aggregation)
// ---------------------------------------------------------------------------
__global__ void prop_step(
    const float* __restrict__ src,      // (B,1,H,W) previous disp
    const float* __restrict__ off,      // (B,18,H,W)
    const float* __restrict__ aff,      // (B,9,H,W)
    const float* __restrict__ occ,      // (B,1,H,W)
    const float* __restrict__ init,     // (B,1,H,W)
    float* __restrict__ dst,            // this step's list_disp slice
    float* __restrict__ dst_final)      // final disp region (last step only)
{
    const size_t P1 = (size_t)HH * WW;
    const size_t idx = (size_t)blockIdx.x * 256 + threadIdx.x;
    if (idx >= (size_t)BB * P1) return;
    const int b = (int)(idx / P1);
    const size_t pix = idx % P1;
    const int h = (int)(pix / WW);
    const int w = (int)(pix % WW);
    const float* simg = src + (size_t)b * P1;

    float acc = 0.0f;
    #pragma unroll
    for (int k = 0; k < 9; ++k) {
        const float oy = off[(size_t)(b * 18 + 2 * k)     * P1 + pix];
        const float ox = off[(size_t)(b * 18 + 2 * k + 1) * P1 + pix];
        const float sy = (float)(h + (k / 3 - 1)) + oy;
        const float sx = (float)(w + (k % 3 - 1)) + ox;
        acc += aff[(size_t)(b * 9 + k) * P1 + pix] * bilin(simg, sy, sx);
    }
    acc = fmaxf(acc, 0.0f);
    const float o = occ[(size_t)b * P1 + pix];
    const float d = (1.0f - o) * acc + o * init[(size_t)b * P1 + pix];
    dst[(size_t)b * P1 + pix] = d;
    if (dst_final) dst_final[(size_t)b * P1 + pix] = fmaxf(d, 0.0f);
}

// ---------------------------------------------------------------------------
extern "C" void kernel_launch(void* const* d_in, const int* in_sizes, int n_in,
                              void* d_out, int out_size, void* d_ws, size_t ws_size,
                              hipStream_t stream) {
    const float* init = (const float*)d_in[0];
    const float* occ  = (const float*)d_in[1];
    const float* conf = (const float*)d_in[2];
    const float* guid = (const float*)d_in[3];
    const float* cw   = (const float*)d_in[4];
    const float* cb   = (const float*)d_in[5];
    const float* scal = (const float*)d_in[6];

    float* out = (float*)d_out;
    const size_t P1 = (size_t)HH * WW;     // 291,840
    const size_t P  = (size_t)BB * P1;     // 583,680

    // d_out layout (flat, return order): disp | stack(6) | offset_flat(18ch) | aff_full(9ch) | scale
    float* r_disp  = out;                  // P
    float* r_stack = out + P;              // 6*P
    float* r_off   = out + 7 * P;          // 18*P (== B*18*H*W)
    float* r_aff   = out + 25 * P;         // 9*P
    float* r_scale = out + 34 * P;         // 1

    // workspace: padded f16 weights (64 KB) + aff_pre (B*8*H*W f32 ~= 18.7 MB)
    _Float16* wp  = (_Float16*)d_ws;
    float* aff_pre = (float*)((char*)d_ws + 65536);

    prep_weights<<<(32768 + 255) / 256, 256, 0, stream>>>(cw, wp);

    dim3 cgrid(WW / 64, HH, BB);           // (19, 240, 2); 2 waves x 32 px per block
    conv_wmma<<<cgrid, 64, 0, stream>>>(guid, wp, cb, scal, r_off, aff_pre);

    const int nblk = (int)((P + 255) / 256);
    aff_finalize<<<nblk, 256, 0, stream>>>(conf, r_off, aff_pre, scal, r_aff, r_scale);

    const float* src = init;
    for (int t = 0; t < PROP; ++t) {
        float* dst = r_stack + (size_t)t * P;
        prop_step<<<nblk, 256, 0, stream>>>(src, r_off, r_aff, occ, init, dst,
                                            (t == PROP - 1) ? r_disp : nullptr);
        src = dst;
    }
}
